// GCN_3822520893971
// MI455X (gfx1250) — compile-verified, matrix operand order checked by reading, not run on hardware
//
#include <hip/hip_runtime.h>

typedef __bf16 bf16_t;
typedef __attribute__((ext_vector_type(16))) __bf16 v16bf;
typedef __attribute__((ext_vector_type(8)))  float  v8f;

#define IN_DIM  512
#define FC_DIM  1024
#define LAT_DIM 512
#define OUT_DIM 256

#define BM 128
#define BN 128
#define BK 32
#define LDS_STRIDE 48   // bf16 elems; 96B rows -> 32B aligned fragment loads

// ---------------- CDNA5 async copy helpers (inline asm) ----------------
// INST_OFFSET is added to BOTH the LDS and global address (ISA 15.18.3),
// so one pair moves 32 contiguous bytes global -> LDS without touching VGPRs.
static __device__ __forceinline__ void async_copy_32B(unsigned lds_addr, const void* gaddr) {
    asm volatile("global_load_async_to_lds_b128 %0, %1, off\n\t"
                 "global_load_async_to_lds_b128 %0, %1, off offset:16"
                 :: "v"(lds_addr), "v"(gaddr) : "memory");
}
static __device__ __forceinline__ void wait_async0() {
    asm volatile("s_wait_asynccnt 0x0" ::: "memory");
}

// ---------------- elementwise helpers ----------------

__global__ void k_cast_f32_bf16(const float* __restrict__ s, bf16_t* __restrict__ d, long long n) {
    long long i = ((long long)blockIdx.x * blockDim.x + threadIdx.x) * 4;
    if (i + 3 < n) {
        float4 v = *(const float4*)(s + i);
        d[i + 0] = (bf16_t)v.x; d[i + 1] = (bf16_t)v.y;
        d[i + 2] = (bf16_t)v.z; d[i + 3] = (bf16_t)v.w;
    } else {
        for (; i < n; ++i) d[i] = (bf16_t)s[i];
    }
}

// weights: s is [K x N] row-major f32 -> d is [N x K] row-major bf16
__global__ void k_cast_transpose_bf16(const float* __restrict__ s, bf16_t* __restrict__ d,
                                      int K, int N) {
    int n = blockIdx.x * blockDim.x + threadIdx.x;
    int k = blockIdx.y;
    if (n < N) d[(size_t)n * K + k] = (bf16_t)s[(size_t)k * N + n];
}

__global__ void k_init_deg(float* deg, int n) {
    int i = blockIdx.x * blockDim.x + threadIdx.x;
    if (i < n) deg[i] = 1.0f;   // self loop
}

__global__ void k_count_deg(const long long* __restrict__ dst, float* deg, int e) {
    int i = blockIdx.x * blockDim.x + threadIdx.x;
    if (i < e) atomicAdd(&deg[(int)dst[i]], 1.0f);
}

__global__ void k_dinv(float* deg, int n) {
    int i = blockIdx.x * blockDim.x + threadIdx.x;
    if (i < n) deg[i] = rsqrtf(deg[i]);   // deg >= 1 always
}

// A[i,f] = dinv[i]^2 * T[i,f] (+ bias[f])   (self-loop term + optional bias)
__global__ void k_self_init(const float* __restrict__ T, const float* __restrict__ dinv,
                            const float* __restrict__ bias, float* __restrict__ A, int F) {
    int i = blockIdx.y;
    int f = blockIdx.x * blockDim.x + threadIdx.x;
    float di = dinv[i];
    float v = di * di * T[(size_t)i * F + f];
    if (bias) v += bias[f];
    A[(size_t)i * F + f] = v;
}

// one block per edge: A[dst, f] += dinv[s]*dinv[d] * T[src, f]
__global__ void k_edge_agg(const float* __restrict__ T, const float* __restrict__ dinv,
                           const long long* __restrict__ src, const long long* __restrict__ dst,
                           float* __restrict__ A, int F) {
    int e = blockIdx.x;
    int s = (int)src[e];
    int d = (int)dst[e];
    float norm = dinv[s] * dinv[d];
    int f = threadIdx.x;
    atomicAdd(&A[(size_t)d * F + f], norm * T[(size_t)s * F + f]);
}

// H[i,f] = bf16(relu(A[i,f] + b[f]))
__global__ void k_relu_bias_cast(const float* __restrict__ A, const float* __restrict__ b,
                                 bf16_t* __restrict__ H, int F) {
    int i = blockIdx.y;
    int f = blockIdx.x * blockDim.x + threadIdx.x;
    float v = A[(size_t)i * F + f] + b[f];
    v = v > 0.0f ? v : 0.0f;
    H[(size_t)i * F + f] = (bf16_t)v;
}

// ---------------- WMMA bf16 GEMM: C[M,N] = A[M,K] * Bt[N,K]^T ----------------
// A is [M x K] row-major bf16, Bt is [N x K] row-major bf16 (pre-transposed weights).
// Double-buffered LDS; tiles staged with global_load_async_to_lds_b128 (ASYNCcnt),
// overlapping the DMA of tile k+1 with WMMA on tile k.
// OUT_MODE 0: f32 store (raw).  OUT_MODE 1: bf16 store of relu(acc + bias[col]).
template <int OUT_MODE>
__global__ __launch_bounds__(256)
void k_wmma_gemm(const bf16_t* __restrict__ A, const bf16_t* __restrict__ Bt,
                 void* __restrict__ C, const float* __restrict__ bias,
                 int M, int N, int K) {
    __shared__ bf16_t As[2][BM * LDS_STRIDE];   // row-major MxK tile
    __shared__ bf16_t Bs[2][BN * LDS_STRIDE];   // n-major:   Bs[n][k]

    const int tid   = threadIdx.x;
    const int wave  = tid >> 5;      // 8 waves
    const int lane  = tid & 31;
    const int waveM = wave & 3;      // 4 slabs of 32 rows
    const int waveN = wave >> 2;     // 2 slabs of 64 cols
    const int l16   = lane & 15;
    const int lh    = lane >> 4;     // 0/1 -> K-half selector per WMMA layout

    const int blockM = blockIdx.x * BM;
    const int blockN = blockIdx.y * BN;

    // staging: 256 threads x 32B per tile; same index map for A and B
    const int srow = tid >> 1;            // 0..127 (row of A tile / n of B tile)
    const int skos = (tid & 1) * 16;      // 0 / 16 (K segment)
    const int grA  = blockM + srow;
    const unsigned ldsOff = (unsigned)(srow * LDS_STRIDE + skos) * 2u;

    v8f acc[2][4];
    const v8f vzero = {0.f, 0.f, 0.f, 0.f, 0.f, 0.f, 0.f, 0.f};
#pragma unroll
    for (int mt = 0; mt < 2; ++mt)
#pragma unroll
        for (int nt = 0; nt < 4; ++nt) acc[mt][nt] = vzero;

    const unsigned asBase[2] = {(unsigned)(uintptr_t)&As[0][0], (unsigned)(uintptr_t)&As[1][0]};
    const unsigned bsBase[2] = {(unsigned)(uintptr_t)&Bs[0][0], (unsigned)(uintptr_t)&Bs[1][0]};

    // M-edge rows: pre-fill both A buffers with zeros once; async issue is
    // EXEC-masked off for those lanes every iteration.
    if (grA >= M) {
        v16bf z;
#pragma unroll
        for (int i = 0; i < 16; ++i) z[i] = (bf16_t)0.0f;
        *(v16bf*)(&As[0][srow * LDS_STRIDE + skos]) = z;
        *(v16bf*)(&As[1][srow * LDS_STRIDE + skos]) = z;
    }

    const int nK = K / BK;

    // prologue: stage tile 0 into buffer 0
    if (grA < M)
        async_copy_32B(asBase[0] + ldsOff, A + (size_t)grA * K + skos);
    async_copy_32B(bsBase[0] + ldsOff, Bt + (size_t)(blockN + srow) * K + skos);
    wait_async0();
    __syncthreads();

    for (int kt = 0; kt < nK; ++kt) {
        const int cur = kt & 1;
        // stage next tile into the other buffer (overlaps with WMMA below)
        if (kt + 1 < nK) {
            const int k1 = (kt + 1) * BK;
            if (grA < M)
                async_copy_32B(asBase[cur ^ 1] + ldsOff, A + (size_t)grA * K + k1 + skos);
            async_copy_32B(bsBase[cur ^ 1] + ldsOff, Bt + (size_t)(blockN + srow) * K + k1 + skos);
        }

        // fragments: lanes 0-15 carry K 0-15, lanes 16-31 carry K 16-31
        v16bf afrag[2], bfrag[4];
#pragma unroll
        for (int mt = 0; mt < 2; ++mt) {
            int row = waveM * 32 + mt * 16 + l16;
            afrag[mt] = *(const v16bf*)(&As[cur][row * LDS_STRIDE + lh * 16]);
        }
#pragma unroll
        for (int nt = 0; nt < 4; ++nt) {
            int col = waveN * 64 + nt * 16 + l16;
            bfrag[nt] = *(const v16bf*)(&Bs[cur][col * LDS_STRIDE + lh * 16]);
        }
#pragma unroll
        for (int mt = 0; mt < 2; ++mt)
#pragma unroll
            for (int nt = 0; nt < 4; ++nt)
                acc[mt][nt] = __builtin_amdgcn_wmma_f32_16x16x32_bf16(
                    false, afrag[mt], false, bfrag[nt],
                    (short)0, acc[mt][nt], false, false);

        wait_async0();      // next tile fully landed in LDS
        __syncthreads();    // all waves done reading 'cur' before it is reused
    }

    // store: VGPR j -> row j (lanes 0-15) / row 8+j (lanes 16-31), col = l16
#pragma unroll
    for (int mt = 0; mt < 2; ++mt) {
#pragma unroll
        for (int nt = 0; nt < 4; ++nt) {
            int col = blockN + waveN * 64 + nt * 16 + l16;
            int rb  = blockM + waveM * 32 + mt * 16 + lh * 8;
#pragma unroll
            for (int j = 0; j < 8; ++j) {
                int r = rb + j;
                if (r < M) {
                    float v = acc[mt][nt][j];
                    if (OUT_MODE == 1) {
                        v += bias[col];
                        v = v > 0.0f ? v : 0.0f;
                        ((bf16_t*)C)[(size_t)r * N + col] = (bf16_t)v;
                    } else {
                        ((float*)C)[(size_t)r * N + col] = v;
                    }
                }
            }
        }
    }
}

// ---------------- driver ----------------

extern "C" void kernel_launch(void* const* d_in, const int* in_sizes, int n_in,
                              void* d_out, int out_size, void* d_ws, size_t ws_size,
                              hipStream_t stream) {
    const float*     x    = (const float*)d_in[0];
    const long long* ei   = (const long long*)d_in[1];   // int64 per reference
    const float*     fc_w = (const float*)d_in[2];
    const float*     fc_b = (const float*)d_in[3];
    const float*     w1   = (const float*)d_in[4];
    const float*     b1   = (const float*)d_in[5];
    const float*     w2   = (const float*)d_in[6];
    const float*     b2   = (const float*)d_in[7];

    const int Nn = in_sizes[0] / IN_DIM;     // 50000
    const int E  = in_sizes[1] / 2;          // 800000
    const long long* srcE = ei;
    const long long* dstE = ei + E;

    // workspace carve-out (256B aligned)
    char* ws = (char*)d_ws;
    size_t off = 0;
    auto carve = [&](size_t bytes) -> void* {
        void* p = ws + off;
        off = (off + bytes + 255) & ~(size_t)255;
        return p;
    };
    bf16_t* WfcT = (bf16_t*)carve((size_t)IN_DIM * FC_DIM * 2);   // [FC_DIM x IN_DIM]
    bf16_t* W1T  = (bf16_t*)carve((size_t)FC_DIM * LAT_DIM * 2);  // [LAT_DIM x FC_DIM]
    bf16_t* W2T  = (bf16_t*)carve((size_t)LAT_DIM * OUT_DIM * 2); // [OUT_DIM x LAT_DIM]
    float*  dinv = (float*)carve((size_t)Nn * 4);
    bf16_t* Xb   = (bf16_t*)carve((size_t)Nn * IN_DIM * 2);       // reused as H2b
    bf16_t* H1b  = (bf16_t*)carve((size_t)Nn * FC_DIM * 2);
    float*  T1   = (float*)carve((size_t)Nn * LAT_DIM * 4);       // reused as T2
    float*  A1   = (float*)carve((size_t)Nn * LAT_DIM * 4);
    bf16_t* H2b  = Xb;
    float*  T2   = T1;
    float*  Out  = (float*)d_out;

    // 1) weight transpose-casts + activation cast to bf16
    k_cast_transpose_bf16<<<dim3(FC_DIM / 256, IN_DIM),  256, 0, stream>>>(fc_w, WfcT, IN_DIM,  FC_DIM);
    k_cast_transpose_bf16<<<dim3(LAT_DIM / 256, FC_DIM), 256, 0, stream>>>(w1,   W1T,  FC_DIM,  LAT_DIM);
    k_cast_transpose_bf16<<<dim3(OUT_DIM / 256, LAT_DIM),256, 0, stream>>>(w2,   W2T,  LAT_DIM, OUT_DIM);
    {
        long long n = (long long)Nn * IN_DIM;
        k_cast_f32_bf16<<<(int)((n / 4 + 255) / 256), 256, 0, stream>>>(x, Xb, n);
    }

    // 2) degrees -> dinv
    k_init_deg<<<(Nn + 255) / 256, 256, 0, stream>>>(dinv, Nn);
    k_count_deg<<<(E + 255) / 256, 256, 0, stream>>>(dstE, dinv, E);
    k_dinv<<<(Nn + 255) / 256, 256, 0, stream>>>(dinv, Nn);

    const int gM = (Nn + BM - 1) / BM;

    // 3) H1 = relu(X @ fc_w + fc_b)            [Nn, FC_DIM] bf16
    k_wmma_gemm<1><<<dim3(gM, FC_DIM / BN), 256, 0, stream>>>(Xb, WfcT, H1b, fc_b, Nn, FC_DIM, IN_DIM);

    // 4) T1 = H1 @ w1                          [Nn, LAT_DIM] f32
    k_wmma_gemm<0><<<dim3(gM, LAT_DIM / BN), 256, 0, stream>>>(H1b, W1T, T1, nullptr, Nn, LAT_DIM, FC_DIM);

    // 5) A1 = dinv^2 * T1 ; scatter edges ; H2 = bf16(relu(A1 + b1))
    k_self_init<<<dim3(LAT_DIM / 256, Nn), 256, 0, stream>>>(T1, dinv, nullptr, A1, LAT_DIM);
    k_edge_agg<<<E, LAT_DIM, 0, stream>>>(T1, dinv, srcE, dstE, A1, LAT_DIM);
    k_relu_bias_cast<<<dim3(LAT_DIM / 256, Nn), 256, 0, stream>>>(A1, b1, H2b, LAT_DIM);

    // 6) T2 = H2 @ w2                          [Nn, OUT_DIM] f32
    k_wmma_gemm<0><<<dim3(gM, OUT_DIM / BN), 256, 0, stream>>>(H2b, W2T, T2, nullptr, Nn, OUT_DIM, LAT_DIM);

    // 7) Out = dinv^2 * T2 + b2 ; scatter edges
    k_self_init<<<dim3(OUT_DIM / 256, Nn), 256, 0, stream>>>(T2, dinv, b2, Out, OUT_DIM);
    k_edge_agg<<<E, OUT_DIM, 0, stream>>>(T2, dinv, srcE, dstE, Out, OUT_DIM);
}
